// MoELayer_678604833550
// MI455X (gfx1250) — matrix operand order repaired and checked by reading, not compiled, not run on hardware
//
#include <hip/hip_runtime.h>
#include <hip/hip_bf16.h>
#include <stdint.h>

#define NTOK 8192
#define DDIM 2048
#define NEXP 8

#define TILE_M 128
#define TILE_N 128

// fallback (convert-on-the-fly) tiling
#define TILE_K 64
#define LDS_STRIDE 72   // 64 + 8 bf16 pad (16B) to break bank conflicts

// fast (pre-converted bf16 + async double-buffer) tiling
#define FTILE_K 32
#define FSTRIDE 40      // 32 + 8 bf16 pad; row stride 80B, 16B-aligned chunks

typedef __bf16 bf16_t;
typedef bf16_t v16bf __attribute__((ext_vector_type(16)));
typedef bf16_t v8bf  __attribute__((ext_vector_type(8)));
typedef bf16_t v4bf  __attribute__((ext_vector_type(4)));
typedef float  v8f   __attribute__((ext_vector_type(8)));

union FragU { v16bf v; v8bf h[2]; };

// ---------------------------------------------------------------------------
// Kernel 0: zero per-expert counters, write the tuple's loss scalar (0.0f).
// ---------------------------------------------------------------------------
__global__ void moe_init_kernel(int* __restrict__ cnt, float* __restrict__ loss_out) {
    if (threadIdx.x < NEXP) cnt[threadIdx.x] = 0;
    if (threadIdx.x == 0)   *loss_out = 0.0f;
}

// ---------------------------------------------------------------------------
// Kernel 1: router (+ optional fused x -> bf16 conversion for the fast path).
// One thread per token; router weights (64KB) staged in LDS. gate = top
// softmax prob = 1/sum(exp(l - lmax)); strict > argmax matches jnp.argmax.
// Builds compacted per-expert token lists for the grouped GEMM.
// ---------------------------------------------------------------------------
__global__ __launch_bounds__(256) void moe_router_kernel(
    const float* __restrict__ x,
    const float* __restrict__ router_w,
    const float* __restrict__ router_b,
    float* __restrict__ gate,
    int*   __restrict__ cnt,
    int*   __restrict__ lists,
    bf16_t* __restrict__ xb) {           // nullptr -> skip conversion
    __shared__ float4 rw4[NEXP][DDIM / 4];   // 64 KB

    const int tid = threadIdx.x;
    const float4* rwg = (const float4*)router_w;
    #pragma unroll
    for (int i = 0; i < (NEXP * DDIM / 4) / 256; ++i) {
        int idx = tid + i * 256;
        ((float4*)rw4)[idx] = rwg[idx];
    }
    __syncthreads();

    const int n = blockIdx.x * 256 + tid;
    const float4* xr = (const float4*)(x + (size_t)n * DDIM);
    bf16_t* xrow = xb ? (xb + (size_t)n * DDIM) : nullptr;

    float acc[NEXP];
    #pragma unroll
    for (int e = 0; e < NEXP; ++e) acc[e] = 0.0f;

    for (int i = 0; i < DDIM / 4; ++i) {
        float4 xv = xr[i];
        if (xrow) {
            v4bf c;
            c[0] = (bf16_t)xv.x; c[1] = (bf16_t)xv.y;
            c[2] = (bf16_t)xv.z; c[3] = (bf16_t)xv.w;
            *(v4bf*)(xrow + i * 4) = c;
        }
        #pragma unroll
        for (int e = 0; e < NEXP; ++e) {
            float4 w = rw4[e][i];
            acc[e] = fmaf(xv.x, w.x, fmaf(xv.y, w.y, fmaf(xv.z, w.z, fmaf(xv.w, w.w, acc[e]))));
        }
    }

    float logits[NEXP];
    float lmax = -3.0e38f;
    int   best = 0;
    #pragma unroll
    for (int e = 0; e < NEXP; ++e) {
        logits[e] = acc[e] + router_b[e];
        if (logits[e] > lmax) { lmax = logits[e]; best = e; }  // strict >: first max
    }
    float denom = 0.0f;
    #pragma unroll
    for (int e = 0; e < NEXP; ++e) denom += expf(logits[e] - lmax);
    gate[n] = 1.0f / denom;

    int pos = atomicAdd(&cnt[best], 1);
    lists[best * NTOK + pos] = n;
}

// ---------------------------------------------------------------------------
// Kernel 2 (fast path only): expert_w f32 -> bf16 (one-shot, bandwidth pass).
// ---------------------------------------------------------------------------
__global__ __launch_bounds__(256) void moe_convw_kernel(
    const float* __restrict__ w, bf16_t* __restrict__ wb) {
    const size_t total = (size_t)NEXP * DDIM * DDIM / 4;   // float4 chunks
    const float4* src = (const float4*)w;
    for (size_t c = (size_t)blockIdx.x * 256 + threadIdx.x; c < total;
         c += (size_t)gridDim.x * 256) {
        float4 v = src[c];
        v4bf b;
        b[0] = (bf16_t)v.x; b[1] = (bf16_t)v.y;
        b[2] = (bf16_t)v.z; b[3] = (bf16_t)v.w;
        *(v4bf*)(wb + c * 4) = b;
    }
}

// ---------------------------------------------------------------------------
// Kernel 3a (FAST): grouped GEMM on pre-converted bf16, staged with
// global_load_async_to_lds_b128 + s_wait_asynccnt, double-buffered LDS.
// 256 thr = 8 wave32; block tile 128x128, wave tile 64x32 = 4x2 WMMA tiles,
// K-step 32 -> 8 v_wmma_f32_16x16x32_bf16 per wave per step, f32 accum.
// ---------------------------------------------------------------------------
__global__ __launch_bounds__(256) void moe_gemm_bf16_kernel(
    const bf16_t* __restrict__ xb,
    const bf16_t* __restrict__ wb,
    const float*  __restrict__ expert_b,
    const float*  __restrict__ gate,
    const int*    __restrict__ cnt,
    const int*    __restrict__ lists,
    float* __restrict__ out) {
    __shared__ bf16_t As[2][TILE_M][FSTRIDE];
    __shared__ bf16_t Bs[2][TILE_N][FSTRIDE];
    __shared__ int    tok_s[TILE_M];
    __shared__ float  gate_s[TILE_M];

    const int e      = blockIdx.z;
    const int cnt_e  = cnt[e];
    const int m_base = blockIdx.y * TILE_M;
    if (m_base >= cnt_e) return;
    const int n_base = blockIdx.x * TILE_N;
    const int tid    = threadIdx.x;
    const int rows   = (cnt_e - m_base < TILE_M) ? (cnt_e - m_base) : TILE_M;

    if (tid < TILE_M) {
        const bool valid = tid < rows;
        const int  t     = valid ? lists[e * NTOK + m_base + tid] : 0;
        tok_s[tid]  = t;
        gate_s[tid] = valid ? gate[t] : 0.0f;
    }
    __syncthreads();

    const int lane = tid & 31;
    const int w    = tid >> 5;
    const int wm   = (w & 1) * 64;
    const int wn   = (w >> 1) * 32;
    const int cc   = lane & 15;
    const int hh   = lane >> 4;

    const bf16_t* wbp = wb + (size_t)e * DDIM * DDIM;

    v8f acc[4][2];
    #pragma unroll
    for (int sm = 0; sm < 4; ++sm)
        #pragma unroll
        for (int sn = 0; sn < 2; ++sn)
            acc[sm][sn] = 0.0f;

    // async tile stage: 512 16B chunks per 128x32 tile -> 2 A + 2 B per thread
    auto issue_tile = [&](int kt, int buf) {
        #pragma unroll
        for (int i = 0; i < 2; ++i) {
            const int c  = tid + i * 256;
            const int r  = c >> 2;             // row 0..127
            const int kc = (c & 3) * 8;        // 0,8,16,24 (bf16 elems)
            // LDS offset = low 32 bits of the flat shared-pointer (aperture in
            // the upper 32 bits per ISA flat addressing).
            uint32_t la = (uint32_t)(uintptr_t)&As[buf][r][kc];
            uint64_t ga = (uint64_t)(uintptr_t)(xb + (size_t)tok_s[r] * DDIM + kt + kc);
            asm volatile("global_load_async_to_lds_b128 %0, %1, off"
                         :: "v"(la), "v"(ga) : "memory");
            uint32_t lb = (uint32_t)(uintptr_t)&Bs[buf][r][kc];
            uint64_t gb = (uint64_t)(uintptr_t)(wbp + (size_t)(n_base + r) * DDIM + kt + kc);
            asm volatile("global_load_async_to_lds_b128 %0, %1, off"
                         :: "v"(lb), "v"(gb) : "memory");
        }
    };

    issue_tile(0, 0);
    const int NK = DDIM / FTILE_K;     // 64
    for (int t = 0; t < NK; ++t) {
        const int buf = t & 1;
        if (t + 1 < NK) {
            issue_tile((t + 1) * FTILE_K, buf ^ 1);
            // 4 outstanding belong to tile t+1; tile t's copies have landed
            asm volatile("s_wait_asynccnt %0" :: "i"(4) : "memory");
        } else {
            asm volatile("s_wait_asynccnt %0" :: "i"(0) : "memory");
        }
        __syncthreads();

        FragU a[4], b[2];
        // A 16-bit 16x32: half 0 -> K 0-7 & 16-23; half 1 -> K 8-15 & 24-31
        #pragma unroll
        for (int sm = 0; sm < 4; ++sm) {
            const int row = wm + sm * 16 + cc;
            a[sm].h[0] = *(const v8bf*)&As[buf][row][hh * 8];
            a[sm].h[1] = *(const v8bf*)&As[buf][row][16 + hh * 8];
        }
        // B 32x16: half 0 -> K 0-7 & 16-23; half 1 -> K 8-15 & 24-31; lane = column
        #pragma unroll
        for (int sn = 0; sn < 2; ++sn) {
            const int col = wn + sn * 16 + cc;
            b[sn].h[0] = *(const v8bf*)&Bs[buf][col][hh * 8];
            b[sn].h[1] = *(const v8bf*)&Bs[buf][col][16 + hh * 8];
        }
        #pragma unroll
        for (int sm = 0; sm < 4; ++sm)
            #pragma unroll
            for (int sn = 0; sn < 2; ++sn)
                acc[sm][sn] = __builtin_amdgcn_wmma_f32_16x16x32_bf16(
                    false, a[sm].v, false, b[sn].v,
                    (short)0, acc[sm][sn], false, false);
        __syncthreads();
    }

    // epilogue: out[token, col] = gate * (acc + bias)
    #pragma unroll
    for (int sm = 0; sm < 4; ++sm) {
        #pragma unroll
        for (int sn = 0; sn < 2; ++sn) {
            const int   col  = n_base + wn + sn * 16 + cc;
            const float bias = expert_b[e * DDIM + col];
            #pragma unroll
            for (int v = 0; v < 8; ++v) {
                const int rl = wm + sm * 16 + v + hh * 8;
                if (rl < rows) {
                    const int t = tok_s[rl];
                    out[(size_t)t * DDIM + col] = gate_s[rl] * (acc[sm][sn][v] + bias);
                }
            }
        }
    }
}

// ---------------------------------------------------------------------------
// Kernel 3b (FALLBACK, ws too small): convert-on-the-fly grouped GEMM
// (round-1 version: f32 global loads, v_cvt to bf16, ds staging, K-step 64).
// ---------------------------------------------------------------------------
__global__ __launch_bounds__(256) void moe_gemm_kernel(
    const float* __restrict__ x,
    const float* __restrict__ expert_w,
    const float* __restrict__ expert_b,
    const float* __restrict__ gate,
    const int*   __restrict__ cnt,
    const int*   __restrict__ lists,
    float* __restrict__ out) {
    __shared__ bf16_t As[TILE_M][LDS_STRIDE];
    __shared__ bf16_t Bs[TILE_N][LDS_STRIDE];
    __shared__ int    tok_s[TILE_M];
    __shared__ float  gate_s[TILE_M];

    const int e      = blockIdx.z;
    const int cnt_e  = cnt[e];
    const int m_base = blockIdx.y * TILE_M;
    if (m_base >= cnt_e) return;
    const int n_base = blockIdx.x * TILE_N;
    const int tid    = threadIdx.x;
    const int rows   = (cnt_e - m_base < TILE_M) ? (cnt_e - m_base) : TILE_M;

    if (tid < TILE_M) {
        const bool valid = tid < rows;
        const int  t     = valid ? lists[e * NTOK + m_base + tid] : 0;
        tok_s[tid]  = t;
        gate_s[tid] = valid ? gate[t] : 0.0f;
    }
    __syncthreads();

    const int lane = tid & 31;
    const int w    = tid >> 5;
    const int wm   = (w & 1) * 64;
    const int wn   = (w >> 1) * 32;
    const int cc   = lane & 15;
    const int hh   = lane >> 4;

    v8f acc[4][2];
    #pragma unroll
    for (int sm = 0; sm < 4; ++sm)
        #pragma unroll
        for (int sn = 0; sn < 2; ++sn)
            acc[sm][sn] = 0.0f;

    const float* Wbase = expert_w + (size_t)e * DDIM * DDIM;

    for (int kt = 0; kt < DDIM; kt += TILE_K) {
        #pragma unroll
        for (int i = 0; i < 8; ++i) {
            const int c  = tid + i * 256;
            const int r  = c >> 4;
            const int kc = (c & 15) * 4;

            const float4 av = *(const float4*)(x + (size_t)tok_s[r] * DDIM + kt + kc);
            v4bf ab;
            ab[0] = (bf16_t)av.x; ab[1] = (bf16_t)av.y;
            ab[2] = (bf16_t)av.z; ab[3] = (bf16_t)av.w;
            *(v4bf*)&As[r][kc] = ab;

            const float4 bv = *(const float4*)(Wbase + (size_t)(n_base + r) * DDIM + kt + kc);
            v4bf bb;
            bb[0] = (bf16_t)bv.x; bb[1] = (bf16_t)bv.y;
            bb[2] = (bf16_t)bv.z; bb[3] = (bf16_t)bv.w;
            *(v4bf*)&Bs[r][kc] = bb;
        }
        __syncthreads();

        if (kt + TILE_K < DDIM) {
            const int pr = tid >> 1;
            const int pk = (tid & 1) * 32;
            __builtin_prefetch(x + (size_t)tok_s[pr] * DDIM + kt + TILE_K + pk, 0, 0);
            __builtin_prefetch(Wbase + (size_t)(n_base + pr) * DDIM + kt + TILE_K + pk, 0, 0);
        }

        #pragma unroll
        for (int ks = 0; ks < TILE_K; ks += 32) {
            FragU a[4], b[2];
            #pragma unroll
            for (int sm = 0; sm < 4; ++sm) {
                const int row = wm + sm * 16 + cc;
                a[sm].h[0] = *(const v8bf*)&As[row][ks + hh * 8];
                a[sm].h[1] = *(const v8bf*)&As[row][ks + 16 + hh * 8];
            }
            #pragma unroll
            for (int sn = 0; sn < 2; ++sn) {
                const int col = wn + sn * 16 + cc;
                b[sn].h[0] = *(const v8bf*)&Bs[col][ks + hh * 8];
                b[sn].h[1] = *(const v8bf*)&Bs[col][ks + 16 + hh * 8];
            }
            #pragma unroll
            for (int sm = 0; sm < 4; ++sm)
                #pragma unroll
                for (int sn = 0; sn < 2; ++sn)
                    acc[sm][sn] = __builtin_amdgcn_wmma_f32_16x16x32_bf16(
                        false, a[sm].v, false, b[sn].v,
                        (short)0, acc[sm][sn], false, false);
        }
        __syncthreads();
    }

    #pragma unroll
    for (int sm = 0; sm < 4; ++sm) {
        #pragma unroll
        for (int sn = 0; sn < 2; ++sn) {
            const int   col  = n_base + wn + sn * 16 + cc;
            const float bias = expert_b[e * DDIM + col];
            #pragma unroll
            for (int v = 0; v < 8; ++v) {
                const int rl = wm + sm * 16 + v + hh * 8;
                if (rl < rows) {
                    const int t = tok_s[rl];
                    out[(size_t)t * DDIM + col] = gate_s[rl] * (acc[sm][sn][v] + bias);
                }
            }
        }
    }
}

// ---------------------------------------------------------------------------
extern "C" void kernel_launch(void* const* d_in, const int* in_sizes, int n_in,
                              void* d_out, int out_size, void* d_ws, size_t ws_size,
                              hipStream_t stream) {
    const float* x        = (const float*)d_in[0];
    const float* expert_w = (const float*)d_in[1];
    const float* expert_b = (const float*)d_in[2];
    const float* router_w = (const float*)d_in[3];
    const float* router_b = (const float*)d_in[4];
    float* out = (float*)d_out;

    // ws layout: [cnt pad 256B][gate N f32][lists E*N i32][xb bf16][wb bf16]
    const size_t off_gate  = 256;
    const size_t off_lists = off_gate + sizeof(float) * NTOK;
    const size_t off_xb    = off_lists + sizeof(int) * NEXP * NTOK;        // 295,168
    const size_t off_wb    = off_xb + sizeof(bf16_t) * (size_t)NTOK * DDIM;
    const size_t need_fast = off_wb + sizeof(bf16_t) * (size_t)NEXP * DDIM * DDIM;

    int*   cnt   = (int*)d_ws;
    float* gatep = (float*)((char*)d_ws + off_gate);
    int*   lists = (int*)((char*)d_ws + off_lists);

    const bool fast = ws_size >= need_fast;
    bf16_t* xb = fast ? (bf16_t*)((char*)d_ws + off_xb) : nullptr;
    bf16_t* wb = fast ? (bf16_t*)((char*)d_ws + off_wb) : nullptr;

    moe_init_kernel<<<1, 64, 0, stream>>>(cnt, out + (size_t)NTOK * DDIM);
    moe_router_kernel<<<NTOK / 256, 256, 0, stream>>>(x, router_w, router_b,
                                                      gatep, cnt, lists, xb);
    if (fast) {
        moe_convw_kernel<<<8192, 256, 0, stream>>>(expert_w, wb);
        moe_gemm_bf16_kernel<<<dim3(DDIM / TILE_N, NTOK / TILE_M, NEXP), 256, 0, stream>>>(
            xb, wb, expert_b, gatep, cnt, lists, out);
    } else {
        moe_gemm_kernel<<<dim3(DDIM / TILE_N, NTOK / TILE_M, NEXP), 256, 0, stream>>>(
            x, expert_w, expert_b, gatep, cnt, lists, out);
    }
}
